// FG_RCA_72198400246157
// MI455X (gfx1250) — matrix operand order repaired and checked by reading, not compiled, not run on hardware
//
#include <hip/hip_runtime.h>
#include <hip/hip_bf16.h>

#define BB 8
#define CC 256
#define HH 128
#define WW 128
#define HW (HH*WW)
#define NGRP 32
#define CPG (CC/NGRP)   // 8 channels per group

typedef __attribute__((ext_vector_type(16))) __bf16        v16bf;
typedef __attribute__((ext_vector_type(8)))  float         v8f;
typedef __attribute__((ext_vector_type(4)))  unsigned int  v4u;

static __device__ __forceinline__ unsigned short f2bf(float f) {
  union { float f; unsigned u; } v; v.f = f;
  unsigned r = v.u + 0x7fffu + ((v.u >> 16) & 1u);   // RNE
  return (unsigned short)(r >> 16);
}
static __device__ __forceinline__ float bf2f(unsigned short h) {
  union { unsigned u; float f; } v; v.u = ((unsigned)h) << 16;
  return v.f;
}

union FragU { v4u q[2]; v16bf v; };
static __device__ __forceinline__ v16bf load_frag(const unsigned short* p) {
  FragU u;
  u.q[0] = *(const v4u*)(p);
  u.q[1] = *(const v4u*)(p + 8);
  return u.v;
}

// ---------------------------------------------------------------------------
// Repack conv weights (co,ci,kh,kw) f32 -> per-lane WMMA A fragments, bf16.
// Layout: [co_tile(16)][tap(9)][chunk(8)][lane(32)][half(16)]
// Lane/K mapping: contiguous, K = 16*(lane/16) + j (B fragments use the same order).
// ---------------------------------------------------------------------------
__global__ __launch_bounds__(256) void repack_w(const float* __restrict__ w,
                                                unsigned short* __restrict__ rw) {
  const int idx = blockIdx.x * 256 + threadIdx.x;
  if (idx >= 16 * 9 * 8 * 32 * 16) return;
  const int j     = idx & 15;
  const int lane  = (idx >> 4) & 31;
  const int chunk = (idx >> 9) & 7;
  const int tmp   = idx >> 12;            // co_tile*9 + tap
  const int tap   = tmp % 9;
  const int cot   = tmp / 9;
  const int m     = lane & 15;
  const int K     = j + ((lane & 16) ? 16 : 0);
  const int co    = cot * 16 + m;
  const int ci    = chunk * 32 + K;
  rw[idx] = f2bf(w[(co * CC + ci) * 9 + tap]);
}

// ---------------------------------------------------------------------------
// 3x3 SAME conv, 256->256, implicit GEMM on v_wmma_f32_16x16x32_bf16.
// Block = 256 thr (8 waves): tile = 128 cout x 64 pixels of one output row.
// Wave (waveM,pg): 4 M-tiles (64 cout) x 16 pixels.
// A-fragment strides (halves): t -> 36864, tap -> 4096, chunk -> 512.
// ---------------------------------------------------------------------------
template<bool LEAKY, bool IN_BF16>
__global__ __launch_bounds__(256) void conv3x3_wmma(
    const void* __restrict__ in_, const unsigned short* __restrict__ rw,
    unsigned short* __restrict__ out)
{
  __shared__ __align__(16) unsigned short sIn[66 * 40];  // [pixel 0..65][ci 0..31], stride 40 halves

  const int tid   = threadIdx.x;
  const int lane  = tid & 31;
  const int wid   = tid >> 5;
  const int waveM = wid >> 2;          // 0..1
  const int pg    = wid & 3;           // 0..3
  const int h     = blockIdx.x >> 1;   // output row
  const int w0    = (blockIdx.x & 1) << 6;
  const int coblk = blockIdx.y;        // 0..1 (128 cout each)
  const int b     = blockIdx.z;

  v8f acc[4] = {};
  const int cotBase = coblk * 8 + waveM * 4;   // co_tile units of 16
  const int khalf   = (lane & 16);             // +0 or +16 halves (K half select)
  const int col     = pg * 16 + (lane & 15);   // output pixel column within tile

  // one base pointer; all fragment offsets become compile-time immediates
  const unsigned short* aBase = rw + (size_t)cotBase * 36864 + (size_t)lane * 16;

#pragma unroll
  for (int kh = 0; kh < 3; ++kh) {
    const int rin = h + kh - 1;
    const bool rValid = (rin >= 0) && (rin < HH);
    for (int chunk = 0; chunk < 8; ++chunk) {
      __syncthreads();
      if (rValid) {
        const int rowBase = ((b * CC + chunk * 32) * HH + rin) * WW;
        for (int i = tid; i < 66 * 32; i += 256) {
          const int ci  = i / 66;
          const int p   = i - ci * 66;
          const int win = w0 - 1 + p;
          unsigned short v = 0;
          if (win >= 0 && win < WW) {
            const int gidx = rowBase + ci * HW + win;
            if (IN_BF16) v = ((const unsigned short*)in_)[gidx];
            else         v = f2bf(((const float*)in_)[gidx]);
          }
          sIn[p * 40 + ci] = v;
        }
        // prefetch next ci-chunk of this input row (global_prefetch path)
        if (chunk < 7 && tid < 96) {
          const int ci  = tid / 3;
          const int seg = tid - ci * 3;
          const size_t e  = (size_t)(rowBase + (32 + ci) * HW + w0);
          const size_t eb = e * (IN_BF16 ? 2u : 4u) + (size_t)seg * 128u;
          __builtin_prefetch((const char*)in_ + eb, 0, 1);
        }
      }
      __syncthreads();
      if (rValid) {
        const unsigned short* aC = aBase + (size_t)chunk * 512;
        // preload B fragments (LDS) and all 12 A fragments (global), then burst WMMA
        const v16bf bfr0 = load_frag(&sIn[(col + 0) * 40 + khalf]);
        const v16bf bfr1 = load_frag(&sIn[(col + 1) * 40 + khalf]);
        const v16bf bfr2 = load_frag(&sIn[(col + 2) * 40 + khalf]);
        v16bf afr[12];
#pragma unroll
        for (int kw = 0; kw < 3; ++kw)
#pragma unroll
          for (int t = 0; t < 4; ++t)
            afr[kw * 4 + t] =
                load_frag(aC + (size_t)t * 36864 + (size_t)(kh * 3 + kw) * 4096);
#pragma unroll
        for (int t = 0; t < 4; ++t)
          acc[t] = __builtin_amdgcn_wmma_f32_16x16x32_bf16(
              false, afr[0 + t], false, bfr0, (short)0, acc[t], false, false);
#pragma unroll
        for (int t = 0; t < 4; ++t)
          acc[t] = __builtin_amdgcn_wmma_f32_16x16x32_bf16(
              false, afr[4 + t], false, bfr1, (short)0, acc[t], false, false);
#pragma unroll
        for (int t = 0; t < 4; ++t)
          acc[t] = __builtin_amdgcn_wmma_f32_16x16x32_bf16(
              false, afr[8 + t], false, bfr2, (short)0, acc[t], false, false);
      }
    }
  }

  // C/D layout: VGPR r -> M = r (+8 for lanes 16..31); N = lane & 15.
  const int wcol = w0 + pg * 16 + (lane & 15);
  const int mAdd = (lane & 16) ? 8 : 0;
#pragma unroll
  for (int t = 0; t < 4; ++t) {
#pragma unroll
    for (int r = 0; r < 8; ++r) {
      const int co = coblk * 128 + waveM * 64 + t * 16 + mAdd + r;
      float v = acc[t][r];
      if (LEAKY) v = (v > 0.f) ? v : 0.1f * v;
      out[(((b * CC + co) * HH + h) * WW) + wcol] = f2bf(v);
    }
  }
}

// ---------------------------------------------------------------------------
// Per-(b,c) spatial mean of bf16 feature map.
// ---------------------------------------------------------------------------
__global__ __launch_bounds__(256) void desc_mean(const unsigned short* __restrict__ feat,
                                                 float* __restrict__ desc) {
  __shared__ float red[256];
  const int bc = blockIdx.x;
  const unsigned int* p = (const unsigned int*)feat + (size_t)bc * (HW / 2);
  float s = 0.f;
  for (int i = threadIdx.x; i < HW / 2; i += 256) {
    const unsigned int u = p[i];
    s += bf2f((unsigned short)(u & 0xffffu)) + bf2f((unsigned short)(u >> 16));
  }
  red[threadIdx.x] = s;
  __syncthreads();
  for (int st = 128; st > 0; st >>= 1) {
    if (threadIdx.x < st) red[threadIdx.x] += red[threadIdx.x + st];
    __syncthreads();
  }
  if (threadIdx.x == 0) desc[bc] = red[0] * (1.f / (float)HW);
}

// ---------------------------------------------------------------------------
// 16x16 DFT -> amp/phase gating -> iDFT -> sigmoid; ECA conv1d; mix weights.
// One block per batch image; 256 threads = one (u,v)/(y,x)/channel each.
// ---------------------------------------------------------------------------
__global__ __launch_bounds__(256) void freq_gate(
    const float* __restrict__ desc, float* __restrict__ mixw,
    const float* amp_w, const float* pha_w,
    const float* freq_t, const float* freq_s,
    const float* hi_gain, const float* lo_gain,
    const float* phs_scale, const float* eca_w,
    const float* mix_alpha, const float* mix_beta)
{
  __shared__ float g[256], sre[256], sim[256], ctab[16], stab[16];
  const int tid = threadIdx.x;
  const int b = blockIdx.x;
  if (tid < 16) {
    float s, c;
    sincosf(6.28318530717958647692f * (float)tid * (1.f / 16.f), &s, &c);
    stab[tid] = s; ctab[tid] = c;
  }
  g[tid] = desc[b * 256 + tid];
  __syncthreads();

  // forward DFT (phases are multiples of 2*pi/16)
  const int u = tid >> 4, vv = tid & 15;
  float re = 0.f, im = 0.f;
  for (int y = 0; y < 16; ++y) {
    const int uy = (u * y) & 15;
    for (int x = 0; x < 16; ++x) {
      const int ph = (uy + vv * x) & 15;
      const float gv = g[y * 16 + x];
      re += gv * ctab[ph];
      im -= gv * stab[ph];
    }
  }
  const float amp = sqrtf(re * re + im * im);
  const float pha = atan2f(im, re);
  const float fy = (u  < 8 ? (float)u  : (float)(u  - 16)) * (1.f / 16.f);
  const float fx = (vv < 8 ? (float)vv : (float)(vv - 16)) * (1.f / 16.f);
  float r = sqrtf(fy * fy + fx * fx) * 1.41421356237309505f;  // / sqrt(0.5)
  r = fminf(r, 1.f);
  const float gate  = 1.f / (1.f + expf(-freq_s[0] * (r - freq_t[0])));
  const float scale = lo_gain[0] * (1.f - gate) + hi_gain[0] * gate;
  const float ga    = amp_w[0] * amp;
  const float gelua = 0.5f * ga * (1.f + erff(ga * 0.70710678118654752f));
  const float amp_adj = amp * (1.f + amp_w[1] * gelua) * scale;
  const float gp    = pha_w[0] * pha;
  const float gelup = 0.5f * gp * (1.f + erff(gp * 0.70710678118654752f));
  const float pha_adj = pha + phs_scale[0] * gate * tanhf(pha_w[1] * gelup);
  sre[tid] = amp_adj * cosf(pha_adj);
  sim[tid] = amp_adj * sinf(pha_adj);
  __syncthreads();

  // inverse DFT, real part
  const int y = tid >> 4, x = tid & 15;
  float acc = 0.f;
  for (int uu = 0; uu < 16; ++uu) {
    const int uy = (uu * y) & 15;
    for (int vx = 0; vx < 16; ++vx) {
      const int ph = (uy + vx * x) & 15;
      acc += sre[uu * 16 + vx] * ctab[ph] - sim[uu * 16 + vx] * stab[ph];
    }
  }
  const float wf = 1.f / (1.f + expf(-acc * (1.f / 256.f)));

  // ECA conv1d k=3, SAME zero pad, over channel index = tid
  const float left  = (tid > 0)   ? g[tid - 1] : 0.f;
  const float right = (tid < 255) ? g[tid + 1] : 0.f;
  const float ye  = eca_w[0] * left + eca_w[1] * g[tid] + eca_w[2] * right;
  const float wec = 1.f / (1.f + expf(-ye));

  const float alpha = 1.f / (1.f + expf(-mix_alpha[0]));
  const float beta  = 1.f / (1.f + expf(-mix_beta[0]));
  mixw[b * 256 + tid] = beta * (alpha * wf + (1.f - alpha) * wec);
}

// ---------------------------------------------------------------------------
// Fused: out = feat*w + x (on the fly) -> depthwise 3x3 -> z; GN stats atomics.
// ---------------------------------------------------------------------------
__global__ __launch_bounds__(256) void dw_gn_stats(
    const unsigned short* __restrict__ feat, const float* __restrict__ x,
    const float* __restrict__ mixw, const float* __restrict__ dw_w,
    float* __restrict__ z, float* __restrict__ stats)
{
  __shared__ float halo[18 * 18];
  const int tid = threadIdx.x;
  const int c = blockIdx.y, b = blockIdx.z;
  const int ty0 = (blockIdx.x >> 3) * 16;
  const int tx0 = (blockIdx.x & 7) * 16;
  const float wv = mixw[b * CC + c];
  const size_t base = (size_t)(b * CC + c) * HW;

  for (int i = tid; i < 18 * 18; i += 256) {
    const int hy = i / 18, hx = i - hy * 18;
    const int gy = ty0 + hy - 1, gx = tx0 + hx - 1;
    float o = 0.f;
    if (gy >= 0 && gy < HH && gx >= 0 && gx < WW) {
      const size_t gi = base + gy * WW + gx;
      o = bf2f(feat[gi]) * wv + x[gi];
    }
    halo[i] = o;
  }
  __syncthreads();
  const int ty = tid >> 4, tx = tid & 15;
  float dz = 0.f;
#pragma unroll
  for (int kh = 0; kh < 3; ++kh)
#pragma unroll
    for (int kw = 0; kw < 3; ++kw)
      dz += dw_w[c * 9 + kh * 3 + kw] * halo[(ty + kh) * 18 + tx + kw];
  z[base + (ty0 + ty) * WW + tx0 + tx] = dz;

  __syncthreads();
  halo[tid] = dz;
  __syncthreads();
  for (int st = 128; st > 0; st >>= 1) { if (tid < st) halo[tid] += halo[tid + st]; __syncthreads(); }
  if (tid == 0) atomicAdd(&stats[(b * NGRP + (c >> 3)) * 2 + 0], halo[0]);
  __syncthreads();
  halo[tid] = dz * dz;
  __syncthreads();
  for (int st = 128; st > 0; st >>= 1) { if (tid < st) halo[tid] += halo[tid + st]; __syncthreads(); }
  if (tid == 0) atomicAdd(&stats[(b * NGRP + (c >> 3)) * 2 + 1], halo[0]);
}

__global__ __launch_bounds__(256) void gn_finalize(
    const float* __restrict__ stats, const float* __restrict__ gn_w,
    const float* __restrict__ gn_b, float* __restrict__ sc, float* __restrict__ bi)
{
  const int i = blockIdx.x * 256 + threadIdx.x;
  if (i >= BB * CC) return;
  const int b = i >> 8, c = i & 255;
  const float n   = (float)(CPG * HW);
  const float sum = stats[(b * NGRP + (c >> 3)) * 2 + 0];
  const float sq  = stats[(b * NGRP + (c >> 3)) * 2 + 1];
  const float mu  = sum / n;
  const float var = sq / n - mu * mu;
  const float rs  = rsqrtf(var + 1e-5f);
  const float s   = gn_w[c] * rs;
  sc[i] = s;
  bi[i] = gn_b[c] - mu * s;
}

__global__ __launch_bounds__(256) void gn_apply_relu(
    const float* __restrict__ z, const float* __restrict__ sc,
    const float* __restrict__ bi, float* __restrict__ out)
{
  const size_t idx = (size_t)blockIdx.x * 256 + threadIdx.x;  // float4 index
  const int bc = (int)(idx >> 12);                            // 4096 float4 per (b,c)
  const float s = sc[bc], t = bi[bc];
  const float4 zv = ((const float4*)z)[idx];
  float4 o;
  o.x = fmaxf(zv.x * s + t, 0.f);
  o.y = fmaxf(zv.y * s + t, 0.f);
  o.z = fmaxf(zv.z * s + t, 0.f);
  o.w = fmaxf(zv.w * s + t, 0.f);
  ((float4*)out)[idx] = o;
}

// ---------------------------------------------------------------------------
extern "C" void kernel_launch(void* const* d_in, const int* in_sizes, int n_in,
                              void* d_out, int out_size, void* d_ws, size_t ws_size,
                              hipStream_t stream)
{
  (void)in_sizes; (void)n_in; (void)out_size; (void)ws_size;
  const float* x       = (const float*)d_in[0];
  const float* c1w     = (const float*)d_in[1];
  const float* c2w     = (const float*)d_in[2];
  const float* amp_w   = (const float*)d_in[3];
  const float* pha_w   = (const float*)d_in[4];
  const float* freq_t  = (const float*)d_in[5];
  const float* freq_s  = (const float*)d_in[6];
  const float* hi_gain = (const float*)d_in[7];
  const float* lo_gain = (const float*)d_in[8];
  const float* phs     = (const float*)d_in[9];
  const float* eca_w   = (const float*)d_in[10];
  const float* mix_a   = (const float*)d_in[11];
  const float* mix_b   = (const float*)d_in[12];
  const float* dw_w    = (const float*)d_in[13];
  const float* gn_w    = (const float*)d_in[14];
  const float* gn_b    = (const float*)d_in[15];

  char* ws = (char*)d_ws;
  size_t off = 0;
  auto carve = [&](size_t bytes) -> void* {
    void* p = ws + off;
    off += (bytes + 255) & ~(size_t)255;
    return p;
  };
  unsigned short* feat1 = (unsigned short*)carve((size_t)BB * CC * HW * 2);
  unsigned short* feat  = (unsigned short*)carve((size_t)BB * CC * HW * 2);
  float*          zbuf  = (float*)carve((size_t)BB * CC * HW * 4);
  unsigned short* rw1   = (unsigned short*)carve((size_t)16 * 9 * 8 * 32 * 16 * 2);
  unsigned short* rw2   = (unsigned short*)carve((size_t)16 * 9 * 8 * 32 * 16 * 2);
  float*          desc  = (float*)carve((size_t)BB * CC * 4);
  float*          mixw  = (float*)carve((size_t)BB * CC * 4);
  float*          stats = (float*)carve((size_t)BB * NGRP * 2 * 4);
  float*          sc    = (float*)carve((size_t)BB * CC * 4);
  float*          bi    = (float*)carve((size_t)BB * CC * 4);

  // 1) repack conv weights into WMMA A-fragment layout (bf16)
  repack_w<<<2304, 256, 0, stream>>>(c1w, rw1);
  repack_w<<<2304, 256, 0, stream>>>(c2w, rw2);

  // 2) conv1 + LeakyReLU(0.1) -> feat1 (bf16)
  conv3x3_wmma<true,  false><<<dim3(256, 2, BB), 256, 0, stream>>>(x,     rw1, feat1);
  // 3) conv2 -> feat (bf16)
  conv3x3_wmma<false, true ><<<dim3(256, 2, BB), 256, 0, stream>>>(feat1, rw2, feat);

  // 4) channel descriptor
  desc_mean<<<BB * CC, 256, 0, stream>>>(feat, desc);

  // 5) DFT gating + ECA + mix -> per-(b,c) weight
  freq_gate<<<BB, 256, 0, stream>>>(desc, mixw, amp_w, pha_w, freq_t, freq_s,
                                    hi_gain, lo_gain, phs, eca_w, mix_a, mix_b);

  // 6) depthwise conv on (feat*w + x) with GN stats
  hipMemsetAsync(stats, 0, (size_t)BB * NGRP * 2 * sizeof(float), stream);
  dw_gn_stats<<<dim3(64, CC, BB), 256, 0, stream>>>(feat, x, mixw, dw_w, zbuf, stats);

  // 7) GroupNorm finalize + apply + ReLU
  gn_finalize<<<(BB * CC + 255) / 256, 256, 0, stream>>>(stats, gn_w, gn_b, sc, bi);
  gn_apply_relu<<<32768, 256, 0, stream>>>(zbuf, sc, bi, (float*)d_out);
}
